// MDPNeuralIRL_49297634623652
// MI455X (gfx1250) — matrix operand order repaired and checked by reading, not compiled, not run on hardware
//
#include <hip/hip_runtime.h>
#include <hip/hip_bf16.h>

typedef __attribute__((ext_vector_type(16))) _Float16 v16h;
typedef __attribute__((ext_vector_type(8)))  _Float16 v8h;
typedef __attribute__((ext_vector_type(8)))  float    v8f;

#define MT   128        // rows per workgroup
#define NH1  256        // hidden width (layers 1,2 out / layer 2,3 in)
#define NH3  128        // layer-3 out width
#define S1   264        // LDS row stride (f16 elems) for 256-wide activations (+8 pad)
#define S3   136        // LDS row stride for 128-wide activations (+8 pad)

// ---------- weight f32 -> f16 conversion (weights stay resident in L2) ----------
__global__ __launch_bounds__(256) void cvt_weights_kernel(
    const float* __restrict__ W2, const float* __restrict__ W3,
    _Float16* __restrict__ W2h, _Float16* __restrict__ W3h)
{
    int i = blockIdx.x * 256 + threadIdx.x;
    if (i < 256 * 256) W2h[i] = (_Float16)W2[i];
    if (i < 128 * 256) W3h[i] = (_Float16)W3[i];
}

__device__ __forceinline__ float silu_f(float x) {
    return x / (1.0f + __expf(-x));
}

// ---------- fused MLP: 7 -> 256 -> 256 -> 128 -> 3 -> softmax ----------
__global__ __launch_bounds__(256) void mlp_fused_kernel(
    const float* __restrict__ log_p,  const float* __restrict__ log_y,
    const float* __restrict__ log_xb, const float* __restrict__ log_q,
    const float* __restrict__ W1, const float* __restrict__ b1,
    const float* __restrict__ b2, const float* __restrict__ b3,
    const float* __restrict__ W4, const float* __restrict__ b4,
    const float* __restrict__ p_log_beta, const float* __restrict__ p_log_delta,
    const _Float16* __restrict__ W2h, const _Float16* __restrict__ W3h,
    float* __restrict__ out)
{
    __shared__ float    xs[MT][8];        // staged 7-wide inputs
    __shared__ _Float16 hA[MT * S1];      // h1 (stride S1); reused for h3 (stride S3)
    __shared__ _Float16 hB[MT * S1];      // h2 (stride S1)

    const int tid  = threadIdx.x;
    const int lane = tid & 31;
    const int wv   = tid >> 5;            // wave id 0..7
    const int m    = lane & 15;           // row (A) / col (B) within 16-tile
    const int sel  = lane >> 4;           // K-half selector per WMMA layout
    const int row0 = blockIdx.x * MT;

    const float beta  = fminf(fmaxf(__expf(p_log_beta[0]), 0.5f), 20.0f);
    const float delta = 1.0f / (1.0f + __expf(-p_log_delta[0]));

    // ---- stage inputs: x = [log_p(3), log_y(1), xb(3)] ----
    for (int i = tid; i < MT * 8; i += 256) {
        int r = i >> 3, c = i & 7;
        int gr = row0 + r;
        float v = 0.0f;
        if (c < 3)       v = log_p[gr * 3 + c];
        else if (c == 3) v = log_y[gr];
        else if (c < 7) {
            int g = c - 4;
            v = delta * log_xb[gr * 3 + g] + (1.0f - delta) * log_q[gr * 3 + g];
        }
        xs[r][c] = v;
    }
    __syncthreads();

    // ---- layer 1 (K=7): VALU, one output column per thread ----
    {
        float w[7];
        #pragma unroll
        for (int k = 0; k < 7; ++k) w[k] = W1[tid * 7 + k];
        const float bb = b1[tid];
        for (int r = 0; r < MT; ++r) {
            float acc = bb;
            #pragma unroll
            for (int k = 0; k < 7; ++k) acc = fmaf(xs[r][k], w[k], acc);
            hA[r * S1 + tid] = (_Float16)silu_f(acc);
        }
    }
    __syncthreads();

    // ---- layer 2: (128x256) x (256x256)^T via v_wmma_f32_16x16x32_f16 ----
    {
        const int   n0    = (wv * 2 + 0) * 16 + m;   // this lane's B column (tile 0)
        const int   n1    = (wv * 2 + 1) * 16 + m;   // this lane's B column (tile 1)
        const float bias0 = b2[n0];
        const float bias1 = b2[n1];
        v8f acc0[MT / 16], acc1[MT / 16];
        #pragma unroll
        for (int mt = 0; mt < MT / 16; ++mt) { v8f z = {}; acc0[mt] = z; acc1[mt] = z; }

        const _Float16* w0 = W2h + n0 * NH1;   // torch-style (out,in): K contiguous
        const _Float16* w1 = W2h + n1 * NH1;

        for (int ks = 0; ks < NH1 / 32; ++ks) {
            const int k0 = ks * 32;
            // B fragment: lanes 0-15 hold K=k0..k0+15, lanes 16-31 hold K=k0+16..k0+31
            v16h bf0 = *(const v16h*)(w0 + k0 + sel * 16);
            v16h bf1 = *(const v16h*)(w1 + k0 + sel * 16);
            #pragma unroll
            for (int mt = 0; mt < MT / 16; ++mt) {
                // A fragment: lanes 0-15 K=0..15, lanes 16-31 K=16..31
                const _Float16* arow = &hA[(mt * 16 + m) * S1 + k0];
                v8h alo = *(const v8h*)(arow + sel * 16);
                v8h ahi = *(const v8h*)(arow + sel * 16 + 8);
                v16h af;
                #pragma unroll
                for (int i = 0; i < 8; ++i) { af[i] = alo[i]; af[i + 8] = ahi[i]; }
                acc0[mt] = __builtin_amdgcn_wmma_f32_16x16x32_f16(
                    false, af, false, bf0, (short)0, acc0[mt], false, false);
                acc1[mt] = __builtin_amdgcn_wmma_f32_16x16x32_f16(
                    false, af, false, bf1, (short)0, acc1[mt], false, false);
            }
        }
        // epilogue: bias + SiLU, write h2 (C layout: vgpr r -> M = r + sel*8, N = m)
        #pragma unroll
        for (int mt = 0; mt < MT / 16; ++mt) {
            const int rb = mt * 16 + sel * 8;
            #pragma unroll
            for (int r = 0; r < 8; ++r) {
                hB[(rb + r) * S1 + n0] = (_Float16)silu_f(acc0[mt][r] + bias0);
                hB[(rb + r) * S1 + n1] = (_Float16)silu_f(acc1[mt][r] + bias1);
            }
        }
    }
    __syncthreads();

    // ---- layer 3: (128x256) x (256x128)^T via WMMA; output into hA (stride S3) ----
    {
        const int   n    = wv * 16 + m;
        const float bias = b3[n];
        v8f acc[MT / 16];
        #pragma unroll
        for (int mt = 0; mt < MT / 16; ++mt) { v8f z = {}; acc[mt] = z; }

        const _Float16* wrow = W3h + n * NH1;
        for (int ks = 0; ks < NH1 / 32; ++ks) {
            const int k0 = ks * 32;
            v16h bf = *(const v16h*)(wrow + k0 + sel * 16);
            #pragma unroll
            for (int mt = 0; mt < MT / 16; ++mt) {
                const _Float16* arow = &hB[(mt * 16 + m) * S1 + k0];
                v8h alo = *(const v8h*)(arow + sel * 16);
                v8h ahi = *(const v8h*)(arow + sel * 16 + 8);
                v16h af;
                #pragma unroll
                for (int i = 0; i < 8; ++i) { af[i] = alo[i]; af[i + 8] = ahi[i]; }
                acc[mt] = __builtin_amdgcn_wmma_f32_16x16x32_f16(
                    false, af, false, bf, (short)0, acc[mt], false, false);
            }
        }
        #pragma unroll
        for (int mt = 0; mt < MT / 16; ++mt) {
            const int rb = mt * 16 + sel * 8;
            #pragma unroll
            for (int r = 0; r < 8; ++r) {
                hA[(rb + r) * S3 + n] = (_Float16)silu_f(acc[mt][r] + bias);
            }
        }
    }
    __syncthreads();

    // ---- layer 4 (128 -> 3) + softmax, one row per thread ----
    if (tid < MT) {
        const _Float16* hrow = &hA[tid * S3];
        const int gr = row0 + tid;
        float l[3];
        #pragma unroll
        for (int g = 0; g < 3; ++g) {
            float acc = b4[g];
            const float* wr = W4 + g * NH3;
            #pragma unroll 8
            for (int k = 0; k < NH3; ++k) acc = fmaf((float)hrow[k], wr[k], acc);
            l[g] = acc * beta;
        }
        float mx = fmaxf(l[0], fmaxf(l[1], l[2]));
        float e0 = __expf(l[0] - mx);
        float e1 = __expf(l[1] - mx);
        float e2 = __expf(l[2] - mx);
        float inv = 1.0f / (e0 + e1 + e2);
        out[gr * 3 + 0] = e0 * inv;
        out[gr * 3 + 1] = e1 * inv;
        out[gr * 3 + 2] = e2 * inv;
    }
}

extern "C" void kernel_launch(void* const* d_in, const int* in_sizes, int n_in,
                              void* d_out, int out_size, void* d_ws, size_t ws_size,
                              hipStream_t stream) {
    const float* log_p     = (const float*)d_in[0];
    const float* log_y     = (const float*)d_in[1];
    const float* log_xb    = (const float*)d_in[2];
    const float* log_q     = (const float*)d_in[3];
    const float* W1        = (const float*)d_in[4];
    const float* b1        = (const float*)d_in[5];
    const float* W2        = (const float*)d_in[6];
    const float* b2        = (const float*)d_in[7];
    const float* W3        = (const float*)d_in[8];
    const float* b3        = (const float*)d_in[9];
    const float* W4        = (const float*)d_in[10];
    const float* b4        = (const float*)d_in[11];
    const float* log_beta  = (const float*)d_in[12];
    const float* log_delta = (const float*)d_in[13];

    _Float16* W2h = (_Float16*)d_ws;            // 256*256 f16 = 128 KB
    _Float16* W3h = W2h + 256 * 256;            // 128*256 f16 =  64 KB

    cvt_weights_kernel<<<256, 256, 0, stream>>>(W2, W3, W2h, W3h);

    const int Brows = in_sizes[1];              // log_y: B elements
    const int grid  = Brows / MT;               // 262144 / 128 = 2048
    mlp_fused_kernel<<<grid, 256, 0, stream>>>(
        log_p, log_y, log_xb, log_q, W1, b1, b2, b3, W4, b4,
        log_beta, log_delta, W2h, W3h, (float*)d_out);
}
